// HyperGen_32375463477385
// MI455X (gfx1250) — matrix operand order repaired and checked by reading, not compiled, not run on hardware
//
#include <hip/hip_runtime.h>
#include <hip/hip_bf16.h>

typedef __attribute__((ext_vector_type(16))) _Float16 v16h;
typedef __attribute__((ext_vector_type(8)))  float    v8f;

#define GHG 8
#define GWG 8

// ---------------- workspace layout (float elements) ----------------
static const size_t OFF_STATS = 0;                        // 1024 floats of stats
static const size_t SZ_X   = (size_t)4 * 37 * 256 * 256;  // 9,699,328 (max B*Cin*H*W)
static const size_t OFF_X  = 1024;
static const size_t OFF_Y  = OFF_X + SZ_X;                // dw output (fp32)
static const size_t OFF_BF = OFF_Y + SZ_X;                // B fragments (f16): max 16,777,216 halves
static const size_t SZ_BF  = (size_t)16777216 / 2;        // in floats
static const size_t OFF_AF = OFF_BF + SZ_BF;              // A fragments (f16): max 1,572,864 halves
static const size_t SZ_AF  = (size_t)1572864 / 2;         // in floats
static const size_t OFF_PW = OFF_AF + SZ_AF;              // pw output (fp32), max 4*21*256*256
static const size_t SZ_PW  = (size_t)4 * 21 * 256 * 256;  // 5,505,024
static const size_t OFF_P  = OFF_PW + SZ_PW;              // normalized p (max 4*32*64*64)

// stats offsets (2 floats per channel: sum, sumsq)
static const int S_DW0 = 0;    // 66 ch
static const int S_PW0 = 132;  // 64 ch
static const int S_DW1 = 260;  // 98 ch
static const int S_PW1 = 456;  // 32 ch
static const int S_DW2 = 520;  // 37 ch
static const int S_PW2 = 594;  // 21 ch

// ---------------- small helpers ----------------
__global__ void k_zero(float* __restrict__ p, int n) {
  int i = blockIdx.x * blockDim.x + threadIdx.x;
  if (i < n) p[i] = 0.f;
}

__device__ __forceinline__ float bilerp(const float* __restrict__ src, long base,
                                        int Hs, int Ws, int Ho, int Wo, int ho, int wo) {
  // jax.image.resize 'bilinear' (half-pixel centers)
  float sy = (float)Hs / (float)Ho;
  float sx = (float)Ws / (float)Wo;
  float fy = ((float)ho + 0.5f) * sy - 0.5f;
  float fx = ((float)wo + 0.5f) * sx - 0.5f;
  float y0f = floorf(fy), x0f = floorf(fx);
  float ty = fy - y0f, tx = fx - x0f;
  int y0 = (int)y0f, x0 = (int)x0f;
  int y1 = y0 + 1, x1 = x0 + 1;
  y0 = y0 < 0 ? 0 : (y0 > Hs - 1 ? Hs - 1 : y0);
  y1 = y1 < 0 ? 0 : (y1 > Hs - 1 ? Hs - 1 : y1);
  x0 = x0 < 0 ? 0 : (x0 > Ws - 1 ? Ws - 1 : x0);
  x1 = x1 < 0 ? 0 : (x1 > Ws - 1 ? Ws - 1 : x1);
  float v00 = src[base + (long)y0 * Ws + x0];
  float v01 = src[base + (long)y0 * Ws + x1];
  float v10 = src[base + (long)y1 * Ws + x0];
  float v11 = src[base + (long)y1 * Ws + x1];
  return (v00 * (1.f - tx) + v01 * tx) * (1.f - ty) +
         (v10 * (1.f - tx) + v11 * tx) * ty;
}

// ---------------- input builders ----------------
__global__ void k_build_x0(const float* __restrict__ f1, float* __restrict__ x, int B) {
  const int C = 66, H = 32, W = 32;
  int idx = blockIdx.x * blockDim.x + threadIdx.x;
  int total = B * C * H * W;
  if (idx >= total) return;
  int w = idx % W; int h = (idx / W) % H; int c = (idx / (W * H)) % C; int b = idx / (W * H * C);
  float v;
  if (c == 0)      v = -1.f + 2.f * (float)w / (float)(W - 1);
  else if (c == 1) v = -1.f + 2.f * (float)h / (float)(H - 1);
  else             v = f1[((b * 64 + (c - 2)) * H + h) * W + w];
  x[idx] = v;
}

__global__ void k_build_x1(const float* __restrict__ f0, const float* __restrict__ p0,
                           float* __restrict__ x, int B) {
  const int C = 98, H = 64, W = 64;
  int idx = blockIdx.x * blockDim.x + threadIdx.x;
  int total = B * C * H * W;
  if (idx >= total) return;
  int w = idx % W; int h = (idx / W) % H; int c = (idx / (W * H)) % C; int b = idx / (W * H * C);
  float v;
  if (c == 0)       v = -1.f + 2.f * (float)w / (float)(W - 1);
  else if (c == 1)  v = -1.f + 2.f * (float)h / (float)(H - 1);
  else if (c < 34)  v = f0[((b * 32 + (c - 2)) * H + h) * W + w];
  else              v = bilerp(p0, (long)(b * 64 + (c - 34)) * 32 * 32, 32, 32, H, W, h, w);
  x[idx] = v;
}

__global__ void k_build_x2(const float* __restrict__ img, const float* __restrict__ p1,
                           float* __restrict__ x, int B) {
  const int C = 37, H = 256, W = 256;
  int idx = blockIdx.x * blockDim.x + threadIdx.x;
  int total = B * C * H * W;
  if (idx >= total) return;
  int w = idx % W; int h = (idx / W) % H; int c = (idx / (W * H)) % C; int b = idx / (W * H * C);
  float v;
  if (c == 0)       v = -1.f + 2.f * (float)w / (float)(W - 1);
  else if (c == 1)  v = -1.f + 2.f * (float)h / (float)(H - 1);
  else if (c < 5)   v = img[((b * 3 + (c - 2)) * H + h) * W + w];
  else              v = bilerp(p1, (long)(b * 32 + (c - 5)) * 64 * 64, 64, 64, H, W, h, w);
  x[idx] = v;
}

// ---------------- depthwise 3x3 with per-patch weights, reflect pad ----------------
__global__ void k_dw3(const float* __restrict__ x, const float* __restrict__ wsig,
                      float* __restrict__ y, int B, int C, int H, int W, int Wch) {
  int idx = blockIdx.x * blockDim.x + threadIdx.x;
  int total = B * C * H * W;
  if (idx >= total) return;
  int w = idx % W; int h = (idx / W) % H; int c = (idx / (W * H)) % C; int b = idx / (W * H * C);
  int p = H / GHG, q = W / GWG;
  int u = h / p, v = w / q;
  long wbase = (((long)b * Wch + (long)c * 9) * GHG + u) * GWG + v;
  long xplane = (long)(b * C + c) * H * W;
  float acc = 0.f;
#pragma unroll
  for (int di = 0; di < 3; ++di) {
    int hh = h + di - 1;
    hh = hh < 0 ? -hh : (hh > H - 1 ? 2 * (H - 1) - hh : hh);
#pragma unroll
    for (int dj = 0; dj < 3; ++dj) {
      int ww = w + dj - 1;
      ww = ww < 0 ? -ww : (ww > W - 1 ? 2 * (W - 1) - ww : ww);
      float wt = wsig[wbase + (long)(di * 3 + dj) * (GHG * GWG)];
      acc += wt * x[xplane + (long)hh * W + ww];
    }
  }
  y[idx] = acc;
}

// ---------------- per-channel sum / sumsq ----------------
__global__ void k_chan_reduce(const float* __restrict__ y, float* __restrict__ stats,
                              int C, int HW) {
  int c = blockIdx.x; int b = blockIdx.y;
  const float* p = y + (long)(b * C + c) * HW;
  float s = 0.f, ss = 0.f;
  for (int i = threadIdx.x; i < HW; i += blockDim.x) {
    float v = p[i]; s += v; ss += v * v;
  }
  __shared__ float sh_s[256];
  __shared__ float sh_q[256];
  sh_s[threadIdx.x] = s; sh_q[threadIdx.x] = ss;
  __syncthreads();
  for (int off = blockDim.x / 2; off > 0; off >>= 1) {
    if ((int)threadIdx.x < off) {
      sh_s[threadIdx.x] += sh_s[threadIdx.x + off];
      sh_q[threadIdx.x] += sh_q[threadIdx.x + off];
    }
    __syncthreads();
  }
  if (threadIdx.x == 0) {
    atomicAdd(&stats[2 * c],     sh_s[0]);
    atomicAdd(&stats[2 * c + 1], sh_q[0]);
  }
}

// ---------------- BN + ReLU6 -> f16 B-fragments (WMMA-ready, zero-padded K) ----------------
// B 32x16 f16 fragment layout: lane L holds column N = L%16; K = (e&7) + 16*(e>=8) + 8*(L/16).
// Fragment buffer: [b][patch][tn][kt][lane][e]  (halves)
__global__ void k_bfrag(const float* __restrict__ y, const float* __restrict__ stats,
                        const float* __restrict__ gam, const float* __restrict__ bet,
                        _Float16* __restrict__ bfr, int B, int Cin, int H, int W,
                        int p, int q, int tilesN, int nKt) {
  int idx = blockIdx.x * blockDim.x + threadIdx.x;
  int total = B * 64 * tilesN * nKt * 512;
  if (idx >= total) return;
  int e    = idx & 15;
  int lane = (idx >> 4) & 31;
  int rest = idx >> 9;
  int kt = rest % nKt;   rest /= nKt;
  int tn = rest % tilesN; rest /= tilesN;
  int patch = rest & 63;
  int b = rest >> 6;
  int grp = lane >> 4;
  int npix = tn * 16 + (lane & 15);
  int kb = kt * 32 + (e & 7) + ((e >> 3) << 4) + grp * 8;
  float v = 0.f;
  if (kb < Cin) {
    int u = patch >> 3, vv = patch & 7;
    int pi = npix / q, pj = npix % q;
    float raw = y[(((long)b * Cin + kb) * H + (u * p + pi)) * W + (vv * q + pj)];
    float N = (float)B * (float)(H * W);
    float m = stats[2 * kb] / N;
    float var = stats[2 * kb + 1] / N - m * m;
    float inv = rsqrtf(var + 1e-5f);
    v = (raw - m) * inv * gam[kb] + bet[kb];
    v = fminf(fmaxf(v, 0.f), 6.f);
  }
  bfr[idx] = (_Float16)v;
}

// ---------------- pw weights -> f16 A-fragments (zero-padded K and M) ----------------
// A 16x32 f16 fragment layout: lane L holds M = L%16; K = (e&7) + 16*(e>=8) + 8*(L/16).
// Fragment buffer: [b][patch][tm][kt][lane][e]  (halves)
__global__ void k_afrag(const float* __restrict__ wsig, _Float16* __restrict__ afr,
                        int B, int Cin, int Cout, int Wch, int wOff,
                        int tilesM, int nKt) {
  int idx = blockIdx.x * blockDim.x + threadIdx.x;
  int total = B * 64 * tilesM * nKt * 512;
  if (idx >= total) return;
  int e    = idx & 15;
  int lane = (idx >> 4) & 31;
  int rest = idx >> 9;
  int kt = rest % nKt;   rest /= nKt;
  int tm = rest % tilesM; rest /= tilesM;
  int patch = rest & 63;
  int b = rest >> 6;
  int grp = lane >> 4;
  int o = tm * 16 + (lane & 15);
  int ka = kt * 32 + (e & 7) + ((e >> 3) << 4) + grp * 8;
  float v = 0.f;
  if (o < Cout && ka < Cin) {
    int u = patch >> 3, vv = patch & 7;
    v = wsig[(((long)b * Wch + wOff + (long)o * Cin + ka) * GHG + u) * GWG + vv];
  }
  afr[idx] = (_Float16)v;
}

// ---------------- BN -> f32 ----------------
__global__ void k_bn_f32(const float* __restrict__ y, const float* __restrict__ stats,
                         const float* __restrict__ gam, const float* __restrict__ bet,
                         float* __restrict__ out, int B, int C, int HW) {
  int idx = blockIdx.x * blockDim.x + threadIdx.x;
  int total = B * C * HW;
  if (idx >= total) return;
  int c = (idx / HW) % C;
  float N = (float)B * (float)HW;
  float m = stats[2 * c] / N;
  float var = stats[2 * c + 1] / N - m * m;
  float inv = rsqrtf(var + 1e-5f);
  out[idx] = (y[idx] - m) * inv * gam[c] + bet[c];
}

// ---------------- pointwise per-patch GEMM via WMMA on pre-swizzled fragments ----------------
// One wave per 16x16 output tile; K loop = unconditional v16h loads + wmma.
__global__ void k_pw_wmma(const _Float16* __restrict__ afr, const _Float16* __restrict__ bfr,
                          float* __restrict__ y, int Cout, int H, int W, int p, int q,
                          int tilesM, int tilesN, int nKt) {
  int lane = threadIdx.x;
  int tm = blockIdx.x % tilesM;
  int tn = blockIdx.x / tilesM;
  int patch = blockIdx.y;
  int b = blockIdx.z;
  const v16h* Ap = (const v16h*)afr + ((long)((b * 64 + patch) * tilesM + tm) * nKt) * 32 + lane;
  const v16h* Bp = (const v16h*)bfr + ((long)((b * 64 + patch) * tilesN + tn) * nKt) * 32 + lane;
  v8f acc = {};
  for (int kt = 0; kt < nKt; ++kt) {
    v16h af = Ap[kt * 32];
    v16h bf = Bp[kt * 32];
    acc = __builtin_amdgcn_wmma_f32_16x16x32_f16(
        /*neg_a=*/false, af, /*neg_b=*/false, bf,
        /*c_mod=*/(short)0, acc, /*reuse_a=*/false, /*reuse_b=*/false);
  }
  // C/D layout: lane L holds col N = L%16, rows M = r + 8*(L/16), r = 0..7
  int u = patch >> 3, v = patch & 7;
  int grp = lane >> 4;
  int npix = tn * 16 + (lane & 15);           // PQ is a multiple of 16 at every level
  int pi = npix / q, pj = npix % q;
  long outpix = (long)(u * p + pi) * W + (v * q + pj);
#pragma unroll
  for (int r = 0; r < 8; ++r) {
    int o = tm * 16 + r + grp * 8;
    if (o < Cout)
      y[((long)b * Cout + o) * H * W + outpix] = acc[r];
  }
}

// ---------------- host ----------------
static inline int cdiv(int a, int b) { return (a + b - 1) / b; }

extern "C" void kernel_launch(void* const* d_in, const int* in_sizes, int n_in,
                              void* d_out, int out_size, void* d_ws, size_t ws_size,
                              hipStream_t stream) {
  (void)in_sizes; (void)n_in; (void)out_size; (void)ws_size;
  const float* x_img = (const float*)d_in[0];
  const float* f0    = (const float*)d_in[1];
  const float* f1    = (const float*)d_in[2];
  const float* w0    = (const float*)d_in[3];
  const float* w1    = (const float*)d_in[4];
  const float* w2    = (const float*)d_in[5];
  const float* g_dw0 = (const float*)d_in[6];  const float* b_dw0 = (const float*)d_in[7];
  const float* g_pw0 = (const float*)d_in[8];  const float* b_pw0 = (const float*)d_in[9];
  const float* g_dw1 = (const float*)d_in[10]; const float* b_dw1 = (const float*)d_in[11];
  const float* g_pw1 = (const float*)d_in[12]; const float* b_pw1 = (const float*)d_in[13];
  const float* g_dw2 = (const float*)d_in[14]; const float* b_dw2 = (const float*)d_in[15];
  const float* g_pw2 = (const float*)d_in[16]; const float* b_pw2 = (const float*)d_in[17];

  float* ws    = (float*)d_ws;
  float* stats = ws + OFF_STATS;
  float* xbuf  = ws + OFF_X;
  float* ybuf  = ws + OFF_Y;
  _Float16* bfrag = (_Float16*)(ws + OFF_BF);
  _Float16* afrag = (_Float16*)(ws + OFF_AF);
  float* pwbuf = ws + OFF_PW;
  float* pbuf  = ws + OFF_P;

  const int B = 4;
  k_zero<<<4, 256, 0, stream>>>(stats, 1024);

  // ================= Level 0: f1 (32x32), Cin=66, Cout=64 =================
  {
    const int C = 66, H = 32, W = 32, Cout = 64, Wch = 4818, wOff = 594, p = 4, q = 4;
    const int HW = H * W, tot = B * C * HW, tot2 = B * Cout * HW;
    const int tM = cdiv(Cout, 16), tN = (p * q) / 16, nKt = cdiv(C, 32);
    k_build_x0<<<cdiv(tot, 256), 256, 0, stream>>>(f1, xbuf, B);
    k_dw3<<<cdiv(tot, 256), 256, 0, stream>>>(xbuf, w0, ybuf, B, C, H, W, Wch);
    k_chan_reduce<<<dim3(C, B), 256, 0, stream>>>(ybuf, stats + S_DW0, C, HW);
    int nb = B * 64 * tN * nKt * 512, na = B * 64 * tM * nKt * 512;
    k_bfrag<<<cdiv(nb, 256), 256, 0, stream>>>(ybuf, stats + S_DW0, g_dw0, b_dw0, bfrag, B, C, H, W, p, q, tN, nKt);
    k_afrag<<<cdiv(na, 256), 256, 0, stream>>>(w0, afrag, B, C, Cout, Wch, wOff, tM, nKt);
    k_pw_wmma<<<dim3(tM * tN, GHG * GWG, B), 32, 0, stream>>>(afrag, bfrag, pwbuf, Cout, H, W, p, q, tM, tN, nKt);
    k_chan_reduce<<<dim3(Cout, B), 256, 0, stream>>>(pwbuf, stats + S_PW0, Cout, HW);
    k_bn_f32<<<cdiv(tot2, 256), 256, 0, stream>>>(pwbuf, stats + S_PW0, g_pw0, b_pw0, pbuf, B, Cout, HW);
  }

  // ================= Level 1: f0 + up(p0) (64x64), Cin=98, Cout=32 =================
  {
    const int C = 98, H = 64, W = 64, Cout = 32, Wch = 4018, wOff = 882, p = 8, q = 8;
    const int HW = H * W, tot = B * C * HW, tot2 = B * Cout * HW;
    const int tM = cdiv(Cout, 16), tN = (p * q) / 16, nKt = cdiv(C, 32);
    k_build_x1<<<cdiv(tot, 256), 256, 0, stream>>>(f0, pbuf, xbuf, B);
    k_dw3<<<cdiv(tot, 256), 256, 0, stream>>>(xbuf, w1, ybuf, B, C, H, W, Wch);
    k_chan_reduce<<<dim3(C, B), 256, 0, stream>>>(ybuf, stats + S_DW1, C, HW);
    int nb = B * 64 * tN * nKt * 512, na = B * 64 * tM * nKt * 512;
    k_bfrag<<<cdiv(nb, 256), 256, 0, stream>>>(ybuf, stats + S_DW1, g_dw1, b_dw1, bfrag, B, C, H, W, p, q, tN, nKt);
    k_afrag<<<cdiv(na, 256), 256, 0, stream>>>(w1, afrag, B, C, Cout, Wch, wOff, tM, nKt);
    k_pw_wmma<<<dim3(tM * tN, GHG * GWG, B), 32, 0, stream>>>(afrag, bfrag, pwbuf, Cout, H, W, p, q, tM, tN, nKt);
    k_chan_reduce<<<dim3(Cout, B), 256, 0, stream>>>(pwbuf, stats + S_PW1, Cout, HW);
    k_bn_f32<<<cdiv(tot2, 256), 256, 0, stream>>>(pwbuf, stats + S_PW1, g_pw1, b_pw1, pbuf, B, Cout, HW);
  }

  // ================= Level 2: x_img + up(p1) (256x256), Cin=37, Cout=21 =================
  {
    const int C = 37, H = 256, W = 256, Cout = 21, Wch = 1110, wOff = 333, p = 32, q = 32;
    const int HW = H * W, tot = B * C * HW, tot2 = B * Cout * HW;
    const int tM = cdiv(Cout, 16), tN = (p * q) / 16, nKt = cdiv(C, 32);
    k_build_x2<<<cdiv(tot, 256), 256, 0, stream>>>(x_img, pbuf, xbuf, B);
    k_dw3<<<cdiv(tot, 256), 256, 0, stream>>>(xbuf, w2, ybuf, B, C, H, W, Wch);
    k_chan_reduce<<<dim3(C, B), 256, 0, stream>>>(ybuf, stats + S_DW2, C, HW);
    int nb = B * 64 * tN * nKt * 512, na = B * 64 * tM * nKt * 512;
    k_bfrag<<<cdiv(nb, 256), 256, 0, stream>>>(ybuf, stats + S_DW2, g_dw2, b_dw2, bfrag, B, C, H, W, p, q, tN, nKt);
    k_afrag<<<cdiv(na, 256), 256, 0, stream>>>(w2, afrag, B, C, Cout, Wch, wOff, tM, nKt);
    k_pw_wmma<<<dim3(tM * tN, GHG * GWG, B), 32, 0, stream>>>(afrag, bfrag, pwbuf, Cout, H, W, p, q, tM, tN, nKt);
    k_chan_reduce<<<dim3(Cout, B), 256, 0, stream>>>(pwbuf, stats + S_PW2, Cout, HW);
    k_bn_f32<<<cdiv(tot2, 256), 256, 0, stream>>>(pwbuf, stats + S_PW2, g_pw2, b_pw2, (float*)d_out, B, Cout, HW);
  }
}
